// AFSIAttention_82729660056413
// MI455X (gfx1250) — matrix operand order repaired and checked by reading, compiled, not run on hardware
//
#include <hip/hip_runtime.h>
#include <math.h>

typedef __attribute__((ext_vector_type(16))) __bf16 v16bf;
typedef __attribute__((ext_vector_type(8)))  __bf16 v8bf;
typedef __attribute__((ext_vector_type(8)))  float  v8f;
typedef __attribute__((ext_vector_type(4)))  unsigned int v4u;
typedef __attribute__((ext_vector_type(8)))  unsigned int v8u;

union F16x16 { v16bf v; v8bf h[2]; };

#define BQ 16   // batch
#define NQ 1024
#define DIMQ 512
#define HQ 8

// ---------------- small reductions ----------------

__global__ __launch_bounds__(256) void rowmean_kernel(const float* __restrict__ x,
                                                      float* __restrict__ xmean,
                                                      int rows, int C) {
    int wave = threadIdx.x >> 5, lane = threadIdx.x & 31;
    int row = blockIdx.x * 8 + wave;
    if (row >= rows) return;
    const float* p = x + (size_t)row * C;
    float s = 0.f;
    for (int i = lane; i < C; i += 32) s += p[i];
    for (int m = 16; m; m >>= 1) s += __shfl_xor(s, m, 32);
    if (lane == 0) xmean[row] = s / (float)C;
}

__global__ __launch_bounds__(256) void colmean_kernel(const float* __restrict__ x,
                                                      float* __restrict__ xavg,
                                                      int B_, int N_, int C) {
    int idx = blockIdx.x * 256 + threadIdx.x;  // b*C + c
    if (idx >= B_ * C) return;
    int b = idx / C, c = idx % C;
    const float* p = x + (size_t)b * N_ * C + c;
    float s = 0.f;
    for (int n = 0; n < N_; n++) s += p[(size_t)n * C];
    xavg[idx] = s / (float)N_;
}

// ---------------- DCT features + head MLP (per-batch block) ----------------

__global__ __launch_bounds__(256) void dct_head_kernel(const float* __restrict__ xmean,
                                                       const float* __restrict__ xavg,
                                                       const float* __restrict__ h1w,
                                                       const float* __restrict__ h1b,
                                                       const float* __restrict__ h2w,
                                                       const float* __restrict__ h2b,
                                                       float* __restrict__ dct_feat,
                                                       float* __restrict__ scal /* [B][2] */) {
    __shared__ float D[32][32];
    __shared__ float A[1024];
    __shared__ float T[1024];
    __shared__ float hid[128];
    __shared__ float red[256];
    __shared__ float hw[8];
    int b = blockIdx.x, t = threadIdx.x;
    const float PI = 3.14159265358979323846f;
    for (int i = t; i < 1024; i += 256) {
        int k = i >> 5, m = i & 31;
        float v = sqrtf(2.f / 32.f) * __cosf(PI * (2.f * m + 1.f) * (float)k / 64.f);
        if (k == 0) v = sqrtf(1.f / 32.f);
        D[k][m] = v;
    }
    for (int i = t; i < 1024; i += 256) A[i] = xmean[b * 1024 + i];
    __syncthreads();
    // T = D @ A
    for (int i = t; i < 1024; i += 256) {
        int k = i >> 5, n = i & 31;
        float s = 0.f;
        for (int m = 0; m < 32; m++) s += D[k][m] * A[m * 32 + n];
        T[i] = s;
    }
    __syncthreads();
    // dct2 = T @ D^T, clipped, reuse A
    for (int i = t; i < 1024; i += 256) {
        int k = i >> 5, l = i & 31;
        float s = 0.f;
        for (int n = 0; n < 32; n++) s += T[k * 32 + n] * D[l][n];
        A[i] = fminf(10.f, fmaxf(-10.f, s));
    }
    __syncthreads();
    // L2 norm
    float ps = 0.f;
    for (int i = t; i < 1024; i += 256) ps += A[i] * A[i];
    red[t] = ps; __syncthreads();
    for (int s = 128; s; s >>= 1) { if (t < s) red[t] += red[t + s]; __syncthreads(); }
    float inv = 1.f / (sqrtf(red[0]) + 1e-5f);
    __syncthreads();
    float pS = 0.f;
    for (int i = t; i < 1024; i += 256) {
        float v = A[i] * inv;
        dct_feat[b * 1024 + i] = v;
        pS += v;
    }
    red[t] = pS; __syncthreads();
    for (int s = 128; s; s >>= 1) { if (t < s) red[t] += red[t + s]; __syncthreads(); }
    float Sb = red[0];
    // head MLP
    if (t < 128) {
        float s = h1b[t];
        for (int i = 0; i < 512; i++) s += xavg[b * 512 + i] * h1w[i * 128 + t];
        hid[t] = fmaxf(s, 0.f);
    }
    __syncthreads();
    if (t < 8) {
        float s = h2b[t];
        for (int i = 0; i < 128; i++) s += hid[i] * h2w[i * 8 + t];
        hw[t] = s;
    }
    __syncthreads();
    if (t == 0) {
        float c = 0.f;
        for (int i = 0; i < 8; i++) c += hw[i] * hw[i];
        scal[b * 2 + 0] = c;
        scal[b * 2 + 1] = Sb;
    }
}

// ---------------- generic bf16-WMMA GEMM: C = A(f32) @ B(f32) + bias ----------------
// tile BM=128, BN=64, BK=64; 256 threads = 8 waves; wave w owns rows w*16..w*16+15

#define BM 128
#define BN 64
#define BK 64

__global__ __launch_bounds__(256) void gemm_kernel(const float* __restrict__ A,
                                                   const float* __restrict__ Bw,
                                                   const float* __restrict__ bias,
                                                   void* __restrict__ Cout,
                                                   int M, int K, int Nn,
                                                   int store_bf16, int scale_cols, float scale) {
    __shared__ __bf16 As[BM][BK];
    __shared__ __bf16 Bt[BN][BK];   // transposed: Bt[n][k]
    int t = threadIdx.x, lane = t & 31, wave = t >> 5;
    int mtile = blockIdx.y * BM, ntile = blockIdx.x * BN;
    int halfo = (lane >= 16) ? 8 : 0;    // A K-half offset per lane group
    int ksel  = (lane >= 16) ? 16 : 0;   // unused; B also uses halfo
    int l15 = lane & 15;
    int arow = wave * 16 + l15;
    v8f acc[4] = {};
    for (int kt = 0; kt < K; kt += BK) {
        // stage A tile (f32 -> bf16)
        for (int i = 0; i < 8; i++) {
            int idx = t + i * 256;           // 2048 float4
            int r = idx >> 4, c4 = idx & 15;
            float4 v = *(const float4*)(A + (size_t)(mtile + r) * K + kt + c4 * 4);
            __bf16* dst = &As[r][c4 * 4];
            dst[0] = (__bf16)v.x; dst[1] = (__bf16)v.y;
            dst[2] = (__bf16)v.z; dst[3] = (__bf16)v.w;
        }
        // stage B tile transposed (f32 -> bf16)
        for (int i = 0; i < 4; i++) {
            int idx = t + i * 256;           // 1024 float4
            int r = idx >> 4, c4 = idx & 15;
            float4 v = *(const float4*)(Bw + (size_t)(kt + r) * Nn + ntile + c4 * 4);
            Bt[c4 * 4 + 0][r] = (__bf16)v.x;
            Bt[c4 * 4 + 1][r] = (__bf16)v.y;
            Bt[c4 * 4 + 2][r] = (__bf16)v.z;
            Bt[c4 * 4 + 3][r] = (__bf16)v.w;
        }
        __syncthreads();
        for (int ks = 0; ks < 2; ks++) {
            F16x16 af;
            af.h[0] = *(const v8bf*)&As[arow][ks * 32 + halfo];
            af.h[1] = *(const v8bf*)&As[arow][ks * 32 + 16 + halfo];
            for (int c = 0; c < 4; c++) {
                F16x16 bfr;
                const __bf16* bp = &Bt[c * 16 + l15][ks * 32 + halfo];
                bfr.h[0] = *(const v8bf*)bp;
                bfr.h[1] = *(const v8bf*)(bp + 16);
                acc[c] = __builtin_amdgcn_wmma_f32_16x16x32_bf16(
                    false, af.v, false, bfr.v, (short)0, acc[c], false, false);
            }
        }
        __syncthreads();
    }
    // epilogue: C layout row = vgpr + 8*(lane>=16), col = lane&15
    int rbase = mtile + wave * 16 + ((lane >= 16) ? 8 : 0);
    for (int c = 0; c < 4; c++) {
        int col = ntile + c * 16 + l15;
        float bb = bias ? bias[col] : 0.f;
        float sc = (col < scale_cols) ? scale : 1.f;
        for (int r = 0; r < 8; r++) {
            float v = (acc[c][r] + bb) * sc;
            size_t o = (size_t)(rbase + r) * Nn + col;
            if (store_bf16) ((__bf16*)Cout)[o] = (__bf16)v;
            else            ((float*)Cout)[o]  = v;
        }
    }
}

// ---------------- fused flash attention with rank-1 freq mixing ----------------
// block = (b, h, 128-query tile); 8 waves, wave owns 16 query rows; key tiles of 64
// K tile staged by the Tensor Data Mover (tensor_load_to_lds); V tile staged
// transposed manually (TDM cannot transpose); next V tile prefetched.

__global__ __launch_bounds__(256) void attn_kernel(const __bf16* __restrict__ qkv,
                                                   const float* __restrict__ dct,
                                                   const float* __restrict__ scal,
                                                   const float* __restrict__ freqw,
                                                   float* __restrict__ outb) {
    __shared__ __bf16 KV[64][64];        // 8KB: K tile (TDM), then re-staged as V^T
    __shared__ float  S[8][16][64];      // 32KB scores
    __shared__ __bf16 P[8][16][64];      // 16KB probabilities (bf16)
    __shared__ float  m_l[8][16], l_l[8][16], corr_l[8][16];
    __shared__ float  coefq[128];
    __shared__ float  dctk[64];
    int t = threadIdx.x, lane = t & 31, wave = t >> 5;
    int qt = blockIdx.x & 7;
    int h  = (blockIdx.x >> 3) & 7;
    int b  = blockIdx.x >> 6;
    float aw = 1.f / (1.f + __expf(-freqw[0]));
    float cB = scal[b * 2 + 0], Sb = scal[b * 2 + 1];
    // per-query-row freq coefficient: coef = c*dct[n] / max(c*dct[n]*Sb, 1e-5)
    for (int i = t; i < 128; i += 256) {
        float rn = cB * dct[b * 1024 + qt * 128 + i];
        coefq[i] = rn / fmaxf(rn * Sb, 1e-5f);
    }
    if (t < 128) { int w = t >> 4, r = t & 15; m_l[w][r] = -INFINITY; l_l[w][r] = 0.f; }
    int halfo = (lane >= 16) ? 8 : 0;
    int ksel  = (lane >= 16) ? 16 : 0;
    int l15 = lane & 15;
    int qrow = qt * 128 + wave * 16 + l15;
    size_t qbase = ((size_t)(b * 1024 + qrow)) * 1536 + h * 64;  // q section (pre-scaled)
    F16x16 aq[2];
    for (int j = 0; j < 2; j++) {
        aq[j].h[0] = *(const v8bf*)(qkv + qbase + j * 32 + halfo);
        aq[j].h[1] = *(const v8bf*)(qkv + qbase + j * 32 + 16 + halfo);
    }
    // LDS byte offset of KV: generic LDS address low 32 bits == LDS offset (ISA 10.2)
    unsigned kv_lds_off = (unsigned)(unsigned long long)(void*)&KV[0][0];
    v8f acc[4] = {};
    __syncthreads();
    for (int kt = 0; kt < 16; kt++) {
        int key0 = kt * 64;
        // ---- stage K tile KV[key][d] via Tensor Data Mover (wave 0 only) ----
        if (wave == 0) {
            unsigned long long gaddr = (unsigned long long)(size_t)qkv
                + 2ull * (((unsigned long long)(b * 1024 + key0)) * 1536 + 512 + (unsigned)h * 64);
            v4u g0;
            g0.x = 1u;                                       // count=1, user mode
            g0.y = kv_lds_off;                               // lds_addr
            g0.z = (unsigned)(gaddr & 0xFFFFFFFFull);        // global_addr[31:0]
            g0.w = (unsigned)((gaddr >> 32) & 0x1FFFFFFull)  // global_addr[56:32]
                 | 0x80000000u;                              // type=2 (image)
            v8u g1 = {
                0x00010000u,        // workgroup_mask=0, data_size=1 (2 bytes)
                (1536u << 16),      // tensor_dim0 = 1536 (low 16 in [31:16])
                (16384u << 16),     // tensor_dim0 hi=0 | tensor_dim1 = 16384 (lo16)
                (64u << 16),        // tensor_dim1 hi=0 | tile_dim0 = 64
                64u,                // tile_dim1 = 64 | tile_dim2 = 0
                1536u,              // tensor_dim0_stride = 1536 (lo32)
                0u,                 // stride0 hi=0 | tensor_dim1_stride lo=0
                0u };
            asm volatile("tensor_load_to_lds %0, %1" :: "s"(g0), "s"(g1) : "memory");
#if __has_builtin(__builtin_amdgcn_s_wait_tensorcnt)
            __builtin_amdgcn_s_wait_tensorcnt(0);
#else
            asm volatile("s_wait_tensorcnt 0x0" ::: "memory");
#endif
        }
        if (t < 64) dctk[t] = dct[b * 1024 + key0 + t];
        // prefetch next iteration's V tile while the TDM is in flight
        if (kt + 1 < 16) {
            int pr = t >> 2;
            __builtin_prefetch(
                qkv + ((size_t)(b * 1024 + key0 + 64 + pr)) * 1536 + 1024 + h * 64 + (t & 3) * 16,
                0, 0);
        }
        __syncthreads();   // K tile visible to all waves
        // ---- S = q @ k^T (B matrix = K^T, so B[k=d][col=key]; Bt layout == KV[key][d]) ----
        int srow = (lane >= 16) ? 8 : 0;
        for (int c = 0; c < 4; c++) {
            v8f s = {};
            for (int j = 0; j < 2; j++) {
                F16x16 bk;
                const __bf16* bp = &KV[c * 16 + l15][j * 32 + halfo];
                bk.h[0] = *(const v8bf*)bp;
                bk.h[1] = *(const v8bf*)(bp + 16);
                s = __builtin_amdgcn_wmma_f32_16x16x32_bf16(
                    false, aq[j].v, false, bk.v, (short)0, s, false, false);
            }
            for (int r = 0; r < 8; r++) S[wave][srow + r][c * 16 + l15] = s[r];
        }
        // ---- online softmax with freq mix, per wave (own rows) ----
        for (int r = 0; r < 16; r++) {
            float coef = coefq[wave * 16 + r];
            float v0 = S[wave][r][lane];
            float v1 = S[wave][r][lane + 32];
            float f0 = fminf(fmaxf(coef * dctk[lane], 0.f), 1.f);
            float f1 = fminf(fmaxf(coef * dctk[lane + 32], 0.f), 1.f);
            v0 = (1.f - aw) * v0 + aw * f0;
            v1 = (1.f - aw) * v1 + aw * f1;
            float mx = fmaxf(v0, v1);
            for (int s2 = 16; s2; s2 >>= 1) mx = fmaxf(mx, __shfl_xor(mx, s2, 32));
            float mprev = m_l[wave][r];
            float mnew = fmaxf(mprev, mx);
            float p0 = __expf(v0 - mnew), p1 = __expf(v1 - mnew);
            float su = p0 + p1;
            for (int s2 = 16; s2; s2 >>= 1) su += __shfl_xor(su, s2, 32);
            float cr = __expf(mprev - mnew);
            if (lane == 0) {
                m_l[wave][r] = mnew;
                l_l[wave][r] = l_l[wave][r] * cr + su;
                corr_l[wave][r] = cr;
            }
            P[wave][r][lane]      = (__bf16)p0;
            P[wave][r][lane + 32] = (__bf16)p1;
        }
        // rescale running output accumulators
        for (int r = 0; r < 8; r++) {
            float cr = corr_l[wave][srow + r];
            for (int c = 0; c < 4; c++) acc[c][r] *= cr;
        }
        __syncthreads();   // all waves done with K tile
        // ---- re-stage KV as V^T: KV[d][key] ----
        {
            int r = t >> 2, c0 = (t & 3) * 16;
            const __bf16* src = qkv + ((size_t)(b * 1024 + key0 + r)) * 1536 + 1024 + h * 64 + c0;
            for (int u = 0; u < 16; u++) KV[c0 + u][r] = src[u];
        }
        __syncthreads();
        // ---- out += P @ V (A = P 16x64keys from LDS; B[k=key][col=d] => Bt == KV[d][key]) ----
        for (int c = 0; c < 4; c++) {
            for (int j = 0; j < 2; j++) {
                F16x16 ap, bv;
                const __bf16* pp = &P[wave][l15][j * 32];
                ap.h[0] = *(const v8bf*)(pp + halfo);
                ap.h[1] = *(const v8bf*)(pp + 16 + halfo);
                const __bf16* vp = &KV[c * 16 + l15][j * 32 + halfo];
                bv.h[0] = *(const v8bf*)vp;
                bv.h[1] = *(const v8bf*)(vp + 16);
                acc[c] = __builtin_amdgcn_wmma_f32_16x16x32_bf16(
                    false, ap.v, false, bv.v, (short)0, acc[c], false, false);
            }
        }
        __syncthreads();   // done with V tile before next K staging (TDM write)
    }
    // write output in (B,N,H*64) layout (== transpose(0,2,1,3).reshape)
    int srow = (lane >= 16) ? 8 : 0;
    for (int c = 0; c < 4; c++) {
        for (int r = 0; r < 8; r++) {
            float l = fmaxf(l_l[wave][srow + r], 1e-20f);
            float v = acc[c][r] / l;
            size_t o = ((size_t)(b * 1024 + qt * 128 + wave * 16 + srow + r)) * 512 + h * 64 + c * 16 + l15;
            outb[o] = v;
        }
    }
}

// ---------------- launcher ----------------

extern "C" void kernel_launch(void* const* d_in, const int* in_sizes, int n_in,
                              void* d_out, int out_size, void* d_ws, size_t ws_size,
                              hipStream_t stream) {
    const float* x      = (const float*)d_in[0];
    const float* qkv_w  = (const float*)d_in[1];
    const float* qkv_b  = (const float*)d_in[2];
    const float* proj_w = (const float*)d_in[3];
    const float* proj_b = (const float*)d_in[4];
    const float* h1w    = (const float*)d_in[5];
    const float* h1b    = (const float*)d_in[6];
    const float* h2w    = (const float*)d_in[7];
    const float* h2b    = (const float*)d_in[8];
    const float* freqw  = (const float*)d_in[9];

    char* ws = (char*)d_ws;
    // workspace layout (bytes)
    __bf16* qkv     = (__bf16*)(void*)(ws + 0);            // 16384*1536*2 = 50331648
    float* attn_out = (float*)(void*)(ws + 50331648);      // 16384*512*4 = 33554432
    float* xmean    = (float*)(void*)(ws + 83886080);      // 16*1024*4   = 65536
    float* xavg     = (float*)(void*)(ws + 83951616);      // 16*512*4    = 32768
    float* dct      = (float*)(void*)(ws + 83984384);      // 16*1024*4   = 65536
    float* scal     = (float*)(void*)(ws + 84049920);      // 16*2*4

    const int M = BQ * NQ;  // 16384
    rowmean_kernel<<<M / 8, 256, 0, stream>>>(x, xmean, M, DIMQ);
    colmean_kernel<<<(BQ * DIMQ) / 256, 256, 0, stream>>>(x, xavg, BQ, NQ, DIMQ);
    dct_head_kernel<<<BQ, 256, 0, stream>>>(xmean, xavg, h1w, h1b, h2w, h2b, dct, scal);
    // QKV GEMM: store bf16, fold SCALE=1/8 into q columns (cols < 512)
    gemm_kernel<<<dim3(1536 / BN, M / BM), 256, 0, stream>>>(
        x, qkv_w, qkv_b, (void*)qkv, M, DIMQ, 1536, /*bf16*/1, /*scale_cols*/512, 0.125f);
    // fused attention: B*H*(N/128) blocks
    attn_kernel<<<BQ * HQ * (NQ / 128), 256, 0, stream>>>(qkv, dct, scal, freqw, attn_out);
    // projection GEMM: fp32 out to d_out
    gemm_kernel<<<dim3(DIMQ / BN, M / BM), 256, 0, stream>>>(
        attn_out, proj_w, proj_b, d_out, M, DIMQ, DIMQ, /*bf16*/0, /*scale_cols*/0, 1.f);
}
